// GNN13_43877385896290
// MI455X (gfx1250) — compile-verified, matrix operand order checked by reading, not run on hardware
//
#include <hip/hip_runtime.h>

// ---------------------------------------------------------------------------
// GNN edge-conditioned conv for MI455X (gfx1250), wave32 + WMMA bf16.
// Weights staged through LDS (ds_load fragments, batched per K-step),
// activations gathered from global; fp32 accumulation; fp32 atomic scatter.
// WMMA chains use the reuse-A operand-cache hint (same A, same opcode).
// ---------------------------------------------------------------------------

typedef __bf16 bf16;
typedef __attribute__((ext_vector_type(16))) __bf16 v16bf;
typedef __attribute__((ext_vector_type(8)))  __bf16 v8bf;
typedef __attribute__((ext_vector_type(8)))  float  v8f;

#define GV   49412   // vocab
#define GEMB 512
#define GEPD 64
#define GMID 256
#define GOUT 256
#define GN   50000   // nodes
#define GE   400000  // edges
#define GG   64      // graphs
#define GE2  128     // 2*EPD

#define LDS_PITCH 136  // 128 K-elements + 8 pad (keeps 16B alignment, spreads banks)

// first WMMA of a K-step: no reuse; later ones: reuse cached A operand
#define WMMA_BF16_RA0(a, b, c) \
  __builtin_amdgcn_wmma_f32_16x16x32_bf16(false, (a), false, (b), (short)0, (c), false, false)
#define WMMA_BF16_RA1(a, b, c) \
  __builtin_amdgcn_wmma_f32_16x16x32_bf16(false, (a), false, (b), (short)0, (c), true, false)

// ---- WMMA fragment loader -------------------------------------------------
// 16-bit A (16x32) lane layout (ISA 7.12.2): lane l holds row m = l&15.
//   lanes 0-15 : elems 0..7 -> K=kb+0..7,  elems 8..15 -> K=kb+16..23
//   lanes 16-31: elems 0..7 -> K=kb+8..15, elems 8..15 -> K=kb+24..31
// B (32x16) is symmetric with lane l covering column n = l&15, so with
// weights pre-transposed to [dout, K] the same loader serves B (global or LDS).
__device__ __forceinline__ v16bf load_frag(const bf16* __restrict__ row, int kb, int hi) {
  const v8bf lo = *reinterpret_cast<const v8bf*>(row + kb + 8 * hi);
  const v8bf hv = *reinterpret_cast<const v8bf*>(row + kb + 8 * hi + 16);
  v16bf a;
#pragma unroll
  for (int i = 0; i < 8; ++i) { a[i] = lo[i]; a[i + 8] = hv[i]; }
  return a;
}

// One GEMM phase: acc[NF] (16 rows x NF*16 cols) += A(16 x kdim) * Wt^T.
// Wt is [dout, kdim] (pre-transposed). B chunks staged cooperatively in LDS;
// per K-step, B fragments are loaded in batches of 4 ahead of the WMMA chain
// so waits amortize 4:1 and ds_loads overlap earlier WMMAs.
// ALL threads of the (256-thread) block must call this identically.
template <int NF>
__device__ __forceinline__ void gemm_lds_phase(const bf16* __restrict__ arow,
                                               const bf16* __restrict__ Wt,
                                               int kdim, int colBase,
                                               int lane, int tid,
                                               bf16* __restrict__ ldsB,
                                               v8f* __restrict__ acc) {
  const int sel = lane & 15;
  const int hi  = lane >> 4;
  const bf16* bbase = ldsB + sel * LDS_PITCH;  // this lane's B column row
  for (int kc = 0; kc < kdim; kc += 128) {
    __syncthreads();  // previous chunk fully consumed before overwrite
    // stage NF*16 weight rows x 128 K (bf16) into LDS, 16B per transfer
#pragma unroll
    for (int idx = tid; idx < NF * 16 * 16; idx += 256) {
      const int c = idx >> 4;
      const int j = idx & 15;
      const v8bf v = *reinterpret_cast<const v8bf*>(
          Wt + (size_t)(colBase + c) * kdim + kc + j * 8);
      *reinterpret_cast<v8bf*>(ldsB + c * LDS_PITCH + j * 8) = v;
    }
    __syncthreads();
#pragma unroll
    for (int kb = 0; kb < 128; kb += 32) {
      v16bf a = load_frag(arow + kc, kb, hi);
#pragma unroll
      for (int g = 0; g < NF; g += 4) {
        // batch the 4 B-fragment loads ahead of the 4-WMMA chain
        v16bf b0 = load_frag(bbase + (size_t)(g + 0) * 16 * LDS_PITCH, kb, hi);
        v16bf b1 = load_frag(bbase + (size_t)(g + 1) * 16 * LDS_PITCH, kb, hi);
        v16bf b2 = load_frag(bbase + (size_t)(g + 2) * 16 * LDS_PITCH, kb, hi);
        v16bf b3 = load_frag(bbase + (size_t)(g + 3) * 16 * LDS_PITCH, kb, hi);
        acc[g + 0] = (g == 0) ? WMMA_BF16_RA0(a, b0, acc[g + 0])
                              : WMMA_BF16_RA1(a, b0, acc[g + 0]);
        acc[g + 1] = WMMA_BF16_RA1(a, b1, acc[g + 1]);
        acc[g + 2] = WMMA_BF16_RA1(a, b2, acc[g + 2]);
        acc[g + 3] = WMMA_BF16_RA1(a, b3, acc[g + 3]);
      }
    }
  }
}

// ---- elementwise / reformat kernels --------------------------------------
__global__ void cvt_bf16_kernel(const float* __restrict__ in, bf16* __restrict__ out, size_t n) {
  size_t i = (size_t)blockIdx.x * blockDim.x + threadIdx.x;
  if (i < n) out[i] = (bf16)in[i];
}

// in [rows, cols] fp32 -> out [keep, rows] bf16 (transpose + truncate columns)
__global__ void transpose_cvt_kernel(const float* __restrict__ in, bf16* __restrict__ out,
                                     int rows, int cols, int keep) {
  size_t i = (size_t)blockIdx.x * blockDim.x + threadIdx.x;
  size_t total = (size_t)rows * keep;
  if (i >= total) return;
  int c = (int)(i % keep);
  int r = (int)(i / keep);
  out[(size_t)c * rows + r] = (bf16)in[(size_t)r * cols + c];
}

__global__ void gather_rows_cvt_kernel(const float* __restrict__ emb, const int* __restrict__ tok,
                                       bf16* __restrict__ out, int dim, size_t total) {
  size_t i = (size_t)blockIdx.x * blockDim.x + threadIdx.x;
  if (i >= total) return;
  size_t row = i / dim;
  int c = (int)(i % dim);
  out[i] = (bf16)emb[(size_t)tok[row] * dim + c];
}

__global__ void zero_f32_kernel(float* __restrict__ p, size_t n) {
  size_t i = (size_t)blockIdx.x * blockDim.x + threadIdx.x;
  if (i < n) p[i] = 0.f;
}

// xh = bf16(relu(xacc)); xacc = 0 (ready for next layer's atomics)
__global__ void relu_cvt_zero_kernel(float* __restrict__ xacc, bf16* __restrict__ xh, size_t n) {
  size_t i = (size_t)blockIdx.x * blockDim.x + threadIdx.x;
  if (i >= n) return;
  float v = xacc[i];
  xh[i] = (bf16)(v > 0.f ? v : 0.f);
  xacc[i] = 0.f;
}

// ---- edge positional-encoding GEMM: ea = bf16(emb[tok]) @ WpT + b ---------
// DOUT = 64 (NF=4). Writes bf16 into ecat row (stride 128) at colOff,
// optionally mirrored into a second ecat buffer (for the invariant ea2 cols).
__global__ void __launch_bounds__(256)
edge_pe_kernel(const bf16* __restrict__ embh, const int* __restrict__ toks,
               const bf16* __restrict__ WpT, const float* __restrict__ bias,
               bf16* __restrict__ out0, bf16* __restrict__ out1,
               int colOff, int nEdges) {
  __shared__ bf16 ldsB[4 * 16 * LDS_PITCH];
  const int tid  = threadIdx.x;
  const int lane = tid & 31;
  const int wave = tid >> 5;
  const int nTiles = (nEdges + 15) >> 4;
  int etile = blockIdx.x * 8 + wave;
  const bool active = etile < nTiles;
  if (!active) etile = nTiles - 1;
  const int e0 = etile * 16;
  int eA = e0 + (lane & 15);
  if (eA >= nEdges) eA = nEdges - 1;
  const bf16* arow = embh + (size_t)toks[eA] * GEMB;

  v8f acc[4] = {};
  gemm_lds_phase<4>(arow, WpT, GEMB, 0, lane, tid, ldsB, acc);

  if (!active) return;
  const int sel = lane & 15, hi = lane >> 4;
  if (out1) {
#pragma unroll
    for (int f = 0; f < 4; ++f) {
      int col = f * 16 + sel;
      float bv = bias[col];
#pragma unroll
      for (int r = 0; r < 8; ++r) {
        int e = e0 + r + 8 * hi;
        if (e < nEdges) {
          bf16 v = (bf16)(acc[f][r] + bv);
          out0[(size_t)e * GE2 + colOff + col] = v;
          out1[(size_t)e * GE2 + colOff + col] = v;
        }
      }
    }
  } else {
#pragma unroll
    for (int f = 0; f < 4; ++f) {
      int col = f * 16 + sel;
      float bv = bias[col];
#pragma unroll
      for (int r = 0; r < 8; ++r) {
        int e = e0 + r + 8 * hi;
        if (e < nEdges) out0[(size_t)e * GE2 + colOff + col] = (bf16)(acc[f][r] + bv);
      }
    }
  }
}

// ---- message GEMM + scatter: msg = xh[src]@WxT + ecat@WeT + bm ------------
// atomically accumulated into xacc[dst]. Wave = 16 edges x 128 cols (NF=8).
__global__ void __launch_bounds__(256)
msg_scatter_kernel(const bf16* __restrict__ xh, int din,
                   const bf16* __restrict__ ecat,
                   const bf16* __restrict__ WxT,  // [dout, din]
                   const bf16* __restrict__ WeT,  // [dout, 128]
                   const float* __restrict__ bm,  // [dout]
                   const int* __restrict__ src, const int* __restrict__ dst,
                   float* __restrict__ xacc, int dout, int nEdges) {
  __shared__ bf16 ldsB[8 * 16 * LDS_PITCH];
  const int tid  = threadIdx.x;
  const int lane = tid & 31;
  const int wave = tid >> 5;
  const int nTiles = (nEdges + 15) >> 4;
  int etile = blockIdx.x * 8 + wave;
  const bool active = etile < nTiles;
  if (!active) etile = nTiles - 1;
  const int e0 = etile * 16;
  const int colBase = blockIdx.y * 128;
  int eA = e0 + (lane & 15);
  if (eA >= nEdges) eA = nEdges - 1;
  const bf16* arow = xh   + (size_t)src[eA] * din;
  const bf16* erow = ecat + (size_t)eA * GE2;

  v8f acc[8] = {};
  gemm_lds_phase<8>(arow, WxT, din, colBase, lane, tid, ldsB, acc);
  gemm_lds_phase<8>(erow, WeT, GE2, colBase, lane, tid, ldsB, acc);

  if (!active) return;
  const int sel = lane & 15, hi = lane >> 4;
#pragma unroll
  for (int f = 0; f < 8; ++f) {
    int col = colBase + f * 16 + sel;
    float bv = bm[col];
#pragma unroll
    for (int r = 0; r < 8; ++r) {
      int e = e0 + r + 8 * hi;
      if (e < nEdges) {
        int d = dst[e];
        atomicAdd(xacc + (size_t)d * dout + col, acc[f][r] + bv);
      }
    }
  }
}

// ---- node GEMM: xacc[n,:] += xh[n,:] @ WrT (NF=8) -------------------------
__global__ void __launch_bounds__(256)
node_gemm_add_kernel(const bf16* __restrict__ xh, int din,
                     const bf16* __restrict__ WrT,  // [dout, din]
                     float* __restrict__ xacc, int dout, int nRows) {
  __shared__ bf16 ldsB[8 * 16 * LDS_PITCH];
  const int tid  = threadIdx.x;
  const int lane = tid & 31;
  const int wave = tid >> 5;
  const int nTiles = (nRows + 15) >> 4;
  int rtile = blockIdx.x * 8 + wave;
  const bool active = rtile < nTiles;
  if (!active) rtile = nTiles - 1;
  const int r0 = rtile * 16;
  const int colBase = blockIdx.y * 128;
  int rA = r0 + (lane & 15);
  if (rA >= nRows) rA = nRows - 1;
  const bf16* arow = xh + (size_t)rA * din;

  v8f acc[8] = {};
  gemm_lds_phase<8>(arow, WrT, din, colBase, lane, tid, ldsB, acc);

  if (!active) return;
  const int sel = lane & 15, hi = lane >> 4;
#pragma unroll
  for (int f = 0; f < 8; ++f) {
    int col = colBase + f * 16 + sel;
#pragma unroll
    for (int r = 0; r < 8; ++r) {
      int row = r0 + r + 8 * hi;
      if (row < nRows) xacc[(size_t)row * dout + col] += acc[f][r];
    }
  }
}

// ---- edge-feature GEMM (only live cols 0..63 of e_new), NF=4 --------------
// e64 = relu(xh[src]@WesT + xh[dst]@WedT + ecat@WeeT + be[:64]) -> bf16 cols 0..63
__global__ void __launch_bounds__(256)
edge_feat_kernel(const bf16* __restrict__ xh, int din,
                 const bf16* __restrict__ ecat,
                 const bf16* __restrict__ WesT,  // [64, din]
                 const bf16* __restrict__ WedT,  // [64, din]
                 const bf16* __restrict__ WeeT,  // [64, 128]
                 const float* __restrict__ be,
                 const int* __restrict__ src, const int* __restrict__ dst,
                 bf16* __restrict__ enext, int nEdges) {
  __shared__ bf16 ldsB[4 * 16 * LDS_PITCH];
  const int tid  = threadIdx.x;
  const int lane = tid & 31;
  const int wave = tid >> 5;
  const int nTiles = (nEdges + 15) >> 4;
  int etile = blockIdx.x * 8 + wave;
  const bool active = etile < nTiles;
  if (!active) etile = nTiles - 1;
  const int e0 = etile * 16;
  int eA = e0 + (lane & 15);
  if (eA >= nEdges) eA = nEdges - 1;
  const bf16* srow = xh   + (size_t)src[eA] * din;
  const bf16* drow = xh   + (size_t)dst[eA] * din;
  const bf16* erow = ecat + (size_t)eA * GE2;

  v8f acc[4] = {};
  gemm_lds_phase<4>(srow, WesT, din, 0, lane, tid, ldsB, acc);
  gemm_lds_phase<4>(drow, WedT, din, 0, lane, tid, ldsB, acc);
  gemm_lds_phase<4>(erow, WeeT, GE2, 0, lane, tid, ldsB, acc);

  if (!active) return;
  const int sel = lane & 15, hi = lane >> 4;
#pragma unroll
  for (int f = 0; f < 4; ++f) {
    int col = f * 16 + sel;
    float bv = be[col];
#pragma unroll
    for (int r = 0; r < 8; ++r) {
      int e = e0 + r + 8 * hi;
      if (e < nEdges) {
        float v = acc[f][r] + bv;
        enext[(size_t)e * GE2 + col] = (bf16)(v > 0.f ? v : 0.f);
      }
    }
  }
}

// ---- final: out[g,:] = xacc[searchsorted(batch, g), :] --------------------
__global__ void master_gather_kernel(const float* __restrict__ xacc,
                                     const int* __restrict__ batch, int n,
                                     float* __restrict__ out) {
  __shared__ int m;
  int g = blockIdx.x;
  if (threadIdx.x == 0) {
    int lo = 0, hi = n;  // first index with batch[i] >= g (side='left')
    while (lo < hi) { int mid = (lo + hi) >> 1; if (batch[mid] < g) lo = mid + 1; else hi = mid; }
    if (lo > n - 1) lo = n - 1;
    m = lo;
  }
  __syncthreads();
  out[(size_t)g * GOUT + threadIdx.x] = xacc[(size_t)m * GOUT + threadIdx.x];
}

// ---------------------------------------------------------------------------
extern "C" void kernel_launch(void* const* d_in, const int* in_sizes, int n_in,
                              void* d_out, int out_size, void* d_ws, size_t ws_size,
                              hipStream_t stream) {
  (void)in_sizes; (void)n_in; (void)out_size; (void)ws_size;

  const int*   x_tokens    = (const int*)d_in[0];
  const int*   edge_index  = (const int*)d_in[1];
  const int*   edge_tokens = (const int*)d_in[2];
  const int*   batch       = (const int*)d_in[3];
  const float* emb1        = (const float*)d_in[5];
  const float* emb2        = (const float*)d_in[6];
  const float* W_pe1 = (const float*)d_in[7];  const float* b_pe1 = (const float*)d_in[8];
  const float* W_pe2 = (const float*)d_in[9];  const float* b_pe2 = (const float*)d_in[10];

  struct Layer { const float *Wx, *We, *bm, *Wr, *Wes, *Wed, *Wee, *be; int din; };
  Layer L[3];
  for (int i = 0; i < 3; ++i) {
    int b = 11 + 8 * i;
    L[i].Wx = (const float*)d_in[b + 0]; L[i].We  = (const float*)d_in[b + 1];
    L[i].bm = (const float*)d_in[b + 2]; L[i].Wr  = (const float*)d_in[b + 3];
    L[i].Wes = (const float*)d_in[b + 4]; L[i].Wed = (const float*)d_in[b + 5];
    L[i].Wee = (const float*)d_in[b + 6]; L[i].be  = (const float*)d_in[b + 7];
    L[i].din = (i == 0) ? GEMB : GMID;
  }

  const int* src = edge_index;        // [E]
  const int* dst = edge_index + GE;   // [E]

  // ---- workspace carve-out (256B aligned) ---------------------------------
  size_t off = 0;
  auto alloc = [&](size_t bytes) -> void* {
    void* p = (char*)d_ws + off;
    off = (off + bytes + 255) & ~(size_t)255;
    return p;
  };
  bf16*  emb1h = (bf16*)alloc((size_t)GV * GEMB * 2);
  bf16*  emb2h = (bf16*)alloc((size_t)GV * GEMB * 2);
  bf16*  xh0   = (bf16*)alloc((size_t)GN * GEMB * 2);   // layer-1 input (bf16)
  bf16*  xh    = (bf16*)alloc((size_t)GN * GMID * 2);   // layer-2/3 input (bf16)
  float* xacc  = (float*)alloc((size_t)GN * GMID * 4);  // node accumulator (fp32)
  bf16*  ecatA = (bf16*)alloc((size_t)GE * GE2 * 2);
  bf16*  ecatB = (bf16*)alloc((size_t)GE * GE2 * 2);
  bf16*  Wpe1t = (bf16*)alloc((size_t)GEPD * GEMB * 2);
  bf16*  Wpe2t = (bf16*)alloc((size_t)GEPD * GEMB * 2);
  bf16 *WxT[3], *WeT[3], *WrT[3], *WesT[2], *WedT[2], *WeeT[2];
  for (int i = 0; i < 3; ++i) {
    WxT[i] = (bf16*)alloc((size_t)GMID * L[i].din * 2);
    WeT[i] = (bf16*)alloc((size_t)GMID * GE2 * 2);
    WrT[i] = (bf16*)alloc((size_t)GMID * L[i].din * 2);
  }
  for (int i = 0; i < 2; ++i) {
    WesT[i] = (bf16*)alloc((size_t)GEPD * L[i].din * 2);
    WedT[i] = (bf16*)alloc((size_t)GEPD * L[i].din * 2);
    WeeT[i] = (bf16*)alloc((size_t)GEPD * GE2 * 2);
  }

  auto blk = [](size_t n) { return (unsigned)((n + 255) / 256); };

  // ---- stage 0: bf16 conversions ------------------------------------------
  cvt_bf16_kernel<<<blk((size_t)GV * GEMB), 256, 0, stream>>>(emb1, emb1h, (size_t)GV * GEMB);
  cvt_bf16_kernel<<<blk((size_t)GV * GEMB), 256, 0, stream>>>(emb2, emb2h, (size_t)GV * GEMB);

  auto tc = [&](const float* in, bf16* out, int rows, int cols, int keep) {
    transpose_cvt_kernel<<<blk((size_t)rows * keep), 256, 0, stream>>>(in, out, rows, cols, keep);
  };
  tc(W_pe1, Wpe1t, GEMB, GEPD, GEPD);
  tc(W_pe2, Wpe2t, GEMB, GEPD, GEPD);
  for (int i = 0; i < 3; ++i) {
    tc(L[i].Wx, WxT[i], L[i].din, GMID, GMID);
    tc(L[i].We, WeT[i], GE2,      GMID, GMID);
    tc(L[i].Wr, WrT[i], L[i].din, GMID, GMID);
  }
  for (int i = 0; i < 2; ++i) {           // layer-3 edge update is dead code
    tc(L[i].Wes, WesT[i], L[i].din, GE2, GEPD);   // keep only live 64 cols
    tc(L[i].Wed, WedT[i], L[i].din, GE2, GEPD);
    tc(L[i].Wee, WeeT[i], GE2,      GE2, GEPD);
  }

  gather_rows_cvt_kernel<<<blk((size_t)GN * GEMB), 256, 0, stream>>>(
      emb1, x_tokens, xh0, GEMB, (size_t)GN * GEMB);
  zero_f32_kernel<<<blk((size_t)GN * GMID), 256, 0, stream>>>(xacc, (size_t)GN * GMID);

  // ---- stage 1: edge attrs: ecatA = [ea | ea2], ecatB cols 64..127 = ea2 --
  const unsigned egrid = (unsigned)((GE / 16 + 7) / 8);  // 16 edges/wave, 8 waves/block
  edge_pe_kernel<<<dim3(egrid, 1), 256, 0, stream>>>(
      emb1h, edge_tokens, Wpe1t, b_pe1, ecatA, (bf16*)nullptr, 0, GE);
  edge_pe_kernel<<<dim3(egrid, 1), 256, 0, stream>>>(
      emb2h, edge_tokens, Wpe2t, b_pe2, ecatA, ecatB, GEPD, GE);

  // ---- stage 2: three conv layers -----------------------------------------
  const unsigned ngrid = (unsigned)(((GN + 15) / 16 + 7) / 8);
  bf16* xin   = xh0;
  bf16* ecur  = ecatA;
  bf16* enext = ecatB;
  for (int i = 0; i < 3; ++i) {
    int din = L[i].din;
    msg_scatter_kernel<<<dim3(egrid, GMID / 128), 256, 0, stream>>>(
        xin, din, ecur, WxT[i], WeT[i], L[i].bm, src, dst, xacc, GMID, GE);
    node_gemm_add_kernel<<<dim3(ngrid, GMID / 128), 256, 0, stream>>>(
        xin, din, WrT[i], xacc, GMID, GN);
    if (i < 2) {
      edge_feat_kernel<<<dim3(egrid, 1), 256, 0, stream>>>(
          xin, din, ecur, WesT[i], WedT[i], WeeT[i], L[i].be, src, dst, enext, GE);
      relu_cvt_zero_kernel<<<blk((size_t)GN * GMID), 256, 0, stream>>>(
          xacc, xh, (size_t)GN * GMID);
      xin = xh;
      bf16* t = ecur; ecur = enext; enext = t;
    }
  }

  // ---- stage 3: master-pool gather (no ReLU after layer 3) ----------------
  master_gather_kernel<<<GG, GOUT, 0, stream>>>(xacc, batch, GN, (float*)d_out);
}
